// Explore_Recommendation_Decoder_23132693856436
// MI455X (gfx1250) — matrix operand order repaired and checked by reading, not compiled, not run on hardware
//
#include <hip/hip_runtime.h>
#include <cmath>

typedef _Float16 v8h  __attribute__((ext_vector_type(8)));
typedef _Float16 v16h __attribute__((ext_vector_type(16)));
typedef float    v8f  __attribute__((ext_vector_type(8)));

#define B_   1024
#define S_   50
#define H_   128
#define V_   100000
#define K2_  256      // 2*H
#define W32_ 3125     // V/32 mask words per row
#define NTB_ 782      // ceil(V/128) N-blocks

// ---------------------------------------------------------------- utilities
__global__ __launch_bounds__(256) void zero_mask_kernel(unsigned* __restrict__ mask) {
    int i = blockIdx.x * 256 + threadIdx.x;   // grid sized exactly: 12500*256 = 1024*3125
    mask[i] = 0u;
}

__global__ __launch_bounds__(256) void scatter_kernel(const int* __restrict__ seq,
                                                      unsigned* __restrict__ mask) {
    int i = blockIdx.x * 256 + threadIdx.x;   // 200*256 = 51200 = B*S exactly
    int b = i / S_;
    int v = seq[i];
    if (v > 0 && v < V_)
        atomicOr(&mask[b * W32_ + (v >> 5)], 1u << (v & 31));
}

// ---------------------------------------------------------------- attention -> cat[B,256]
__global__ __launch_bounds__(128) void attention_kernel(
    const float* __restrict__ all_memory, const float* __restrict__ last_memory,
    const float* __restrict__ Ue_w, const float* __restrict__ Ue_b,
    const float* __restrict__ We_w, const float* __restrict__ We_b,
    const float* __restrict__ Ve_w, const float* __restrict__ Ve_b,
    float* __restrict__ cat) {
    __shared__ _Float16 sUe[H_ * 136];      // row-padded f16 copy of Ue_w (34.8KB)
    __shared__ float    sAM[S_ * H_];       // all_memory[b]  (25.6KB)
    __shared__ float    sRed[S_ * 4];
    __shared__ float    sScore[S_];
    __shared__ float    sSum;

    const int b    = blockIdx.x;
    const int k    = threadIdx.x;           // 0..127, output channel
    const int wave = k >> 5, lane = k & 31;

    // stage Ue_w (f32 -> f16) into LDS, coalesced float4
#pragma unroll
    for (int j = 0; j < 32; ++j) {
        int   i4 = k + 128 * j;             // < 4096 float4
        float4 v = *(const float4*)(Ue_w + (size_t)i4 * 4);
        int   e  = i4 * 4;
        int row = e >> 7, col = e & 127;
        _Float16* p = &sUe[row * 136 + col];
        p[0] = (_Float16)v.x; p[1] = (_Float16)v.y;
        p[2] = (_Float16)v.z; p[3] = (_Float16)v.w;
    }
    // stage all_memory[b]
    for (int j = 0; j < 13; ++j) {
        int i4 = k + 128 * j;
        if (i4 < (S_ * H_ / 4))
            *(float4*)&sAM[i4 * 4] = *(const float4*)(all_memory + (size_t)b * (S_ * H_) + i4 * 4);
    }
    __syncthreads();

    // lm[k] = last_memory[b] . We_w[k,:] + We_b[k]
    const float* lmrow = last_memory + b * H_;
    const float* werow = We_w + (size_t)k * H_;
    float lmk = We_b[k];
    for (int h = 0; h < H_; h += 4) {
        float4 wv = *(const float4*)(werow + h);
        lmk += lmrow[h] * wv.x + lmrow[h + 1] * wv.y + lmrow[h + 2] * wv.z + lmrow[h + 3] * wv.w;
    }
    const float base = lmk + Ue_b[k];
    const float vek  = Ve_w[k];

    const v8h* urow = (const v8h*)&sUe[k * 136];   // k*136 is 8-half aligned
    for (int s = 0; s < S_; ++s) {
        float acc = base;
        const float* am = &sAM[s * H_];
#pragma unroll
        for (int h8 = 0; h8 < 16; ++h8) {
            v8h u = urow[h8];
#pragma unroll
            for (int q = 0; q < 8; ++q) acc += am[h8 * 8 + q] * (float)u[q];
        }
        float prod = tanhf(acc) * vek;
        prod += __shfl_xor(prod, 16, 32);
        prod += __shfl_xor(prod, 8, 32);
        prod += __shfl_xor(prod, 4, 32);
        prod += __shfl_xor(prod, 2, 32);
        prod += __shfl_xor(prod, 1, 32);
        if (lane == 0) sRed[s * 4 + wave] = prod;
    }
    __syncthreads();
    if (k < S_)
        sScore[k] = sRed[k * 4] + sRed[k * 4 + 1] + sRed[k * 4 + 2] + sRed[k * 4 + 3] + Ve_b[0];
    __syncthreads();
    if (k == 0) {  // tiny serial softmax over S=50
        float mx = -3.4e38f;
        for (int s = 0; s < S_; ++s) mx = fmaxf(mx, sScore[s]);
        float sum = 0.f;
        for (int s = 0; s < S_; ++s) { float e = __expf(sScore[s] - mx); sScore[s] = e; sum += e; }
        sSum = sum;
    }
    __syncthreads();
    float inv = 1.f / sSum;
    float ctx = 0.f;
    for (int s = 0; s < S_; ++s) ctx += sScore[s] * sAM[s * H_ + k];
    cat[b * K2_ + k]       = ctx * inv;
    cat[b * K2_ + H_ + k]  = lmrow[k];
}

// ---------------------------------------------------------------- WMMA vocab GEMM + mask + partial softmax
// Block tile: 128(M) x 128(N), 8 waves, each wave = one 16-col strip x 8 M-tiles.
// LDS ~76KB (CDNA5 allows up to 320KB/workgroup).
__global__ __launch_bounds__(256) void vocab_gemm_kernel(
    const float* __restrict__ cat, const float* __restrict__ E_w,
    const unsigned* __restrict__ mask, float* __restrict__ out,
    float* __restrict__ pmax, float* __restrict__ psum) {
    __shared__ _Float16 sA[128 * 264];      // 128 rows of cat as f16, pad to 264 halves (67.6KB)
    __shared__ unsigned sMask[128 * 4];     // bitmask words covering this 128x128 tile
    __shared__ float    sPM[128 * 8];
    __shared__ float    sPS[128 * 8];

    const int tid  = threadIdx.x;
    const int wave = tid >> 5, lane = tid & 31;
    const int m0   = blockIdx.y * 128;
    const int n0   = blockIdx.x * 128;

    // stage A tile (cat f32 -> f16): 8192 float4
#pragma unroll
    for (int j = 0; j < 32; ++j) {
        int f   = tid + 256 * j;            // float4 index, < 8192
        int row = f >> 6;
        int col = (f & 63) << 2;
        float4 v = *(const float4*)(cat + (size_t)(m0 + row) * K2_ + col);
        _Float16* p = &sA[row * 264 + col];
        p[0] = (_Float16)v.x; p[1] = (_Float16)v.y;
        p[2] = (_Float16)v.z; p[3] = (_Float16)v.w;
    }
    // stage mask words (n0 is 128-aligned -> word aligned): 512 words
#pragma unroll
    for (int j = 0; j < 2; ++j) {
        int idx = tid + 256 * j;
        int row = idx >> 2, wi = idx & 3;
        int gw  = (n0 >> 5) + wi;
        sMask[idx] = (gw < W32_) ? mask[(size_t)(m0 + row) * W32_ + gw] : 0u;
    }
    __syncthreads();

    const int  n0w   = n0 + wave * 16;
    const bool valid = (n0w + 16) <= V_;    // V is 16-aligned: strips fully valid or fully out
    const int  ncol  = lane & 15;
    const int  nglob = n0w + ncol;
    const int  nload = valid ? nglob : 0;
    const int  koffB = (lane < 16) ? 0 : 16;  // B frag: 16 contiguous K per lane half
    const int  koffA = (lane < 16) ? 0 : 16;  // A frag: 16 contiguous K per lane half
    const int  rbase = lane & 15;

    v8f acc[8] = {};
#pragma unroll
    for (int kt = 0; kt < 8; ++kt) {
        const float4* ep4 = (const float4*)(E_w + (size_t)nload * K2_ + kt * 32 + koffB);
        float4 e0 = ep4[0], e1 = ep4[1], e2 = ep4[2], e3 = ep4[3];
        v16h bf;
        bf[0]  = (_Float16)e0.x; bf[1]  = (_Float16)e0.y; bf[2]  = (_Float16)e0.z; bf[3]  = (_Float16)e0.w;
        bf[4]  = (_Float16)e1.x; bf[5]  = (_Float16)e1.y; bf[6]  = (_Float16)e1.z; bf[7]  = (_Float16)e1.w;
        bf[8]  = (_Float16)e2.x; bf[9]  = (_Float16)e2.y; bf[10] = (_Float16)e2.z; bf[11] = (_Float16)e2.w;
        bf[12] = (_Float16)e3.x; bf[13] = (_Float16)e3.y; bf[14] = (_Float16)e3.z; bf[15] = (_Float16)e3.w;

        const int ka = kt * 32 + koffA;
#pragma unroll
        for (int mt = 0; mt < 8; ++mt) {
            const _Float16* ap = &sA[(mt * 16 + rbase) * 264 + ka];
            v8h a0 = *(const v8h*)ap;            // 16B aligned: 264 = 33*8
            v8h a1 = *(const v8h*)(ap + 8);
            v16h af = __builtin_shufflevector(a0, a1, 0,1,2,3,4,5,6,7,8,9,10,11,12,13,14,15);
            acc[mt] = __builtin_amdgcn_wmma_f32_16x16x32_f16(
                false, af, false, bf, (short)0, acc[mt], false, false);
        }
    }

    // epilogue: mask -> logits, 16-lane online-softmax partials, store logits
    const int mAdd = (lane >= 16) ? 8 : 0;
#pragma unroll
    for (int mt = 0; mt < 8; ++mt) {
#pragma unroll
        for (int r = 0; r < 8; ++r) {
            int   mloc = mt * 16 + r + mAdd;
            float x    = -1e30f;
            if (valid) {
                x = acc[mt][r];
                unsigned mw = sMask[mloc * 4 + ((wave * 16 + ncol) >> 5)];
                if ((mw >> (nglob & 31)) & 1u) x = -1e30f;
            }
            float rmax = x;
            rmax = fmaxf(rmax, __shfl_xor(rmax, 1, 32));
            rmax = fmaxf(rmax, __shfl_xor(rmax, 2, 32));
            rmax = fmaxf(rmax, __shfl_xor(rmax, 4, 32));
            rmax = fmaxf(rmax, __shfl_xor(rmax, 8, 32));
            float e    = __expf(x - rmax);
            float rsum = e;
            rsum += __shfl_xor(rsum, 1, 32);
            rsum += __shfl_xor(rsum, 2, 32);
            rsum += __shfl_xor(rsum, 4, 32);
            rsum += __shfl_xor(rsum, 8, 32);
            if (valid) out[(size_t)(m0 + mloc) * V_ + nglob] = x;
            if (ncol == 0) { sPM[mloc * 8 + wave] = rmax; sPS[mloc * 8 + wave] = rsum; }
        }
    }
    __syncthreads();
    if (tid < 128) {  // combine the 8 wave-strips -> one partial per (row, 128-col block)
        float M = -3.4e38f, Ssum = 0.f;
#pragma unroll
        for (int w = 0; w < 8; ++w) {
            float m_ = sPM[tid * 8 + w], s_ = sPS[tid * 8 + w];
            if (m_ > M) { Ssum = Ssum * __expf(M - m_) + s_; M = m_; }
            else        { Ssum += s_ * __expf(m_ - M); }
        }
        pmax[(size_t)(m0 + tid) * NTB_ + blockIdx.x] = M;
        psum[(size_t)(m0 + tid) * NTB_ + blockIdx.x] = Ssum;
    }
}

// ---------------------------------------------------------------- per-row combine of 782 partials
__global__ __launch_bounds__(256) void row_combine_kernel(
    const float* __restrict__ pmax, const float* __restrict__ psum,
    float* __restrict__ rowM, float* __restrict__ rowL) {
    __shared__ float sM[256], sS[256];
    int b = blockIdx.x, tid = threadIdx.x;
    float M = -3.4e38f, Ssum = 0.f;
    for (int t = tid; t < NTB_; t += 256) {
        float m_ = pmax[(size_t)b * NTB_ + t], s_ = psum[(size_t)b * NTB_ + t];
        if (m_ > M) { Ssum = Ssum * __expf(M - m_) + s_; M = m_; }
        else        { Ssum += s_ * __expf(m_ - M); }
    }
    sM[tid] = M; sS[tid] = Ssum;
    __syncthreads();
    for (int off = 128; off > 0; off >>= 1) {
        if (tid < off) {
            float ma = sM[tid], sa = sS[tid];
            float mb = sM[tid + off], sb = sS[tid + off];
            float mm = fmaxf(ma, mb);
            sM[tid] = mm;
            sS[tid] = sa * __expf(ma - mm) + sb * __expf(mb - mm);
        }
        __syncthreads();
    }
    if (tid == 0) { rowM[b] = sM[0]; rowL[b] = sS[0]; }
}

// ---------------------------------------------------------------- normalize: out = exp(logit - M)/L
__global__ __launch_bounds__(256) void finalize_kernel(
    float* __restrict__ out, const float* __restrict__ rowM, const float* __restrict__ rowL) {
    int b = blockIdx.y;
    int p = blockIdx.x * 256 + threadIdx.x;   // float4 index within row
    if (p < V_ / 4) {
        float M   = rowM[b];
        float inv = 1.f / rowL[b];
        float4* ptr = (float4*)(out + (size_t)b * V_ + (size_t)p * 4);
        float4 v = *ptr;
        v.x = __expf(v.x - M) * inv;
        v.y = __expf(v.y - M) * inv;
        v.z = __expf(v.z - M) * inv;
        v.w = __expf(v.w - M) * inv;
        *ptr = v;
    }
}

// ---------------------------------------------------------------- launcher
extern "C" void kernel_launch(void* const* d_in, const int* in_sizes, int n_in,
                              void* d_out, int out_size, void* d_ws, size_t ws_size,
                              hipStream_t stream) {
    const float* all_memory  = (const float*)d_in[0];
    const float* last_memory = (const float*)d_in[1];
    const int*   seq_item    = (const int*)d_in[2];
    const float* Ue_w        = (const float*)d_in[3];
    const float* Ue_b        = (const float*)d_in[4];
    const float* We_w        = (const float*)d_in[5];
    const float* We_b        = (const float*)d_in[6];
    const float* Ve_w        = (const float*)d_in[7];
    const float* Ve_b        = (const float*)d_in[8];
    const float* E_w         = (const float*)d_in[9];
    float* out = (float*)d_out;

    char* ws = (char*)d_ws;                             // ~20.3 MB used
    float*    cat   = (float*)(ws + 0);                 //  1,048,576 B  (B x 256)
    unsigned* maskb = (unsigned*)(ws + 1048576);        // 12,800,000 B  (B x V/32 bits)
    float*    pmax  = (float*)(ws + 13848576);          //  3,203,072 B  (B x 782)
    float*    psum  = (float*)(ws + 17051648);          //  3,203,072 B
    float*    rowM  = (float*)(ws + 20254720);          //      4,096 B
    float*    rowL  = (float*)(ws + 20258816);          //      4,096 B

    zero_mask_kernel<<<12500, 256, 0, stream>>>(maskb);
    scatter_kernel<<<200, 256, 0, stream>>>(seq_item, maskb);
    attention_kernel<<<B_, 128, 0, stream>>>(all_memory, last_memory,
                                             Ue_w, Ue_b, We_w, We_b, Ve_w, Ve_b, cat);
    vocab_gemm_kernel<<<dim3(NTB_, 8), 256, 0, stream>>>(cat, E_w, maskb, out, pmax, psum);
    row_combine_kernel<<<B_, 256, 0, stream>>>(pmax, psum, rowM, rowL);
    finalize_kernel<<<dim3(98, B_), 256, 0, stream>>>(out, rowM, rowL);
    (void)in_sizes; (void)n_in; (void)out_size; (void)ws_size;
}
